// RotEncoder2D_78417512891253
// MI455X (gfx1250) — matrix operand order repaired and checked by reading, not compiled, not run on hardware
//
#include <hip/hip_runtime.h>
#include <hip/hip_bf16.h>

typedef __attribute__((ext_vector_type(16))) _Float16 v16h;
typedef __attribute__((ext_vector_type(8)))  float    v8f;

#define Bc 4
#define Rc 4
#define Nc 4096
#define Mc 64
#define Fc 512
#define KNc (3 * Nc)

// f16 padded weight offsets (in halves) inside workspace
#define OFF_FW0 0        // 64  x 32
#define OFF_FW1 2048     // 64  x 64
#define OFF_FW2 6144     // 128 x 64
#define OFF_FW3 14336    // 128 x 160  (concat: ch0-1 = x, ch2-129 = h, pad->160)
#define OFF_GW0 34816    // 256 x 160
#define OFF_GW1 75776    // 256 x 256
#define OFF_GW2 141312   // 512 x 256
#define OFF_GW3 272384   // 512 x 672  (cols 0..129 = gin, 130..159 = 0, 160..671 = h)
#define W16_TOTAL 616448

// ---------------- WMMA fragment helpers (per CDNA5 ISA 7.12.2 layouts) -------

// A matrix 16x32 f16, row m striped across VGPRs of one lane.
// lane l (0..15): M=l, halves cover K = {0..7,16..23} (+8 for lanes 16..31).
__device__ __forceinline__ v16h load_a_frag(const _Float16* __restrict__ W,
                                            int Kpad, int m_base, int k_base) {
  int lane = threadIdx.x & 31;
  int l = lane & 15, hi = lane >> 4;
  const _Float16* row = W + (size_t)(m_base + l) * Kpad + k_base;
  v16h a;
#pragma unroll
  for (int vg = 0; vg < 8; ++vg) {
    int k = ((vg < 4) ? 0 : 16) + 2 * (vg & 3) + hi * 8;
    a[2 * vg]     = row[k];
    a[2 * vg + 1] = row[k + 1];
  }
  return a;
}

// B matrix 32x16 f16: N = lane&15; halves cover K = {0..7,16..23} (+8 for lanes 16..31).
// Activations stored point-major (act[pt*CHpad + ch]) -> two runs of 8 contiguous halves.
__device__ __forceinline__ v16h load_b_frag(const _Float16* act, int CHpad,
                                            int pt_base, int k_base) {
  int lane = threadIdx.x & 31;
  int n = lane & 15, hi = lane >> 4;
  const _Float16* p = act + (size_t)(pt_base + n) * CHpad + k_base + hi * 8;
  v16h b;
#pragma unroll
  for (int j = 0; j < 16; ++j) b[j] = p[(j < 8) ? j : j + 8];
  return b;
}

__device__ __forceinline__ v8f tile_mm(const _Float16* __restrict__ Wp, int Wkpad,
                                       int m_base, const _Float16* act, int CHpad,
                                       int pt_base, int kChunks, int wKoff, int aKoff,
                                       v8f c) {
  for (int kc = 0; kc < kChunks; ++kc) {
    v16h a = load_a_frag(Wp, Wkpad, m_base, wKoff + kc * 32);
    v16h b = load_b_frag(act, CHpad, pt_base, aKoff + kc * 32);
    c = __builtin_amdgcn_wmma_f32_16x16x32_f16(false, a, false, b, (short)0, c,
                                               false, false);
  }
  return c;
}

// D 16x16 f32: element (m = i + hi*8, n = lane&15) in c[i]. Bias + ReLU -> f16 LDS.
__device__ __forceinline__ void store_relu(v8f c, const float* __restrict__ bias,
                                           _Float16* act, int CHpad, int pt_base,
                                           int oc_base, int ch_off) {
  int lane = threadIdx.x & 31;
  int n = lane & 15, hi = lane >> 4;
  _Float16* p = act + (size_t)(pt_base + n) * CHpad + ch_off + oc_base + hi * 8;
#pragma unroll
  for (int i = 0; i < 8; ++i) {
    float v = c[i] + bias[oc_base + hi * 8 + i];
    v = v > 0.f ? v : 0.f;
    p[i] = (_Float16)v;
  }
}

// ---------------- small utility kernels --------------------------------------

__global__ void zerof_kernel(float* p, int n) {
  int i = blockIdx.x * 256 + threadIdx.x;
  if (i < n) p[i] = 0.f;
}

// convert/pad f32 weights to f16. split>0 remaps concat columns for gp_W3.
__global__ void wprep_kernel(const float* __restrict__ W, _Float16* __restrict__ dst,
                             int OC, int IC, int ICpad, int split, int splitPad) {
  int i = blockIdx.x * 256 + threadIdx.x;
  if (i >= OC * ICpad) return;
  int oc = i / ICpad, c = i % ICpad;
  int sc;
  if (split)
    sc = (c < split) ? c : ((c < splitPad) ? -1 : c - (splitPad - split));
  else
    sc = (c < IC) ? c : -1;
  float v = (sc >= 0 && sc < IC) ? W[(size_t)oc * IC + sc] : 0.f;
  dst[i] = (_Float16)v;
}

// top-3 nearest SOM nodes per point + cluster accumulation
__global__ void topk_kernel(const float* __restrict__ x, const float* __restrict__ node,
                            int* __restrict__ minidx, float* __restrict__ csum) {
  int gid = blockIdx.x * blockDim.x + threadIdx.x;
  if (gid >= Bc * Nc) return;
  int b = gid / Nc, n = gid % Nc;
  float x0 = x[(b * 2 + 0) * Nc + n], x1 = x[(b * 2 + 1) * Nc + n];
  float d0 = 3.4e38f, d1 = 3.4e38f, d2 = 3.4e38f;
  int i0 = 0, i1 = 0, i2 = 0;
  for (int m = 0; m < Mc; ++m) {
    float dx = x0 - node[(b * 2 + 0) * Mc + m];
    float dy = x1 - node[(b * 2 + 1) * Mc + m];
    float d = dx * dx + dy * dy;
    if (d < d0)      { d2 = d1; i2 = i1; d1 = d0; i1 = i0; d0 = d; i0 = m; }
    else if (d < d1) { d2 = d1; i2 = i1; d1 = d;  i1 = m; }
    else if (d < d2) { d2 = d;  i2 = m; }
  }
  minidx[b * KNc + 0 * Nc + n] = i0;
  minidx[b * KNc + 1 * Nc + n] = i1;
  minidx[b * KNc + 2 * Nc + n] = i2;
  int idx[3] = {i0, i1, i2};
#pragma unroll
  for (int k = 0; k < 3; ++k) {
    float* s = &csum[(b * Mc + idx[k]) * 3];
    atomicAdd(s + 0, x0);
    atomicAdd(s + 1, x1);
    atomicAdd(s + 2, 1.f);
  }
}

// cluster means -> rotated node positions, plus occupancy flags
__global__ void nodes_kernel(const float* __restrict__ csum, float* __restrict__ noder,
                             int* __restrict__ rowmax) {
  int t = threadIdx.x;  // 256 == Bc*Mc
  int b = t / Mc, m = t % Mc;
  float s0 = csum[(b * Mc + m) * 3 + 0];
  float s1 = csum[(b * Mc + m) * 3 + 1];
  float cnt = csum[(b * Mc + m) * 3 + 2];
  float inv = 1.f / (cnt + 1e-5f);
  float mx = s0 * inv, my = s1 * inv;
  rowmax[b * Mc + m] = (cnt > 0.f) ? 1 : 0;
  for (int r = 0; r < Rc; ++r) {
    float th = 6.283185307179586f * (float)r / (float)Rc;
    float ct = __cosf(th), st = __sinf(th);
    noder[(((b * Rc) + r) * 2 + 0) * Mc + m] = ct * mx - st * my;
    noder[(((b * Rc) + r) * 2 + 1) * Mc + m] = st * mx + ct * my;
  }
}

// ---------------- fused fp PointResNet + scatter-max -------------------------
// grid: (KN/64, B*R), block 256 (8 waves). 64-point tile, WMMA GEMMs via LDS.
__global__ __launch_bounds__(256) void fp_kernel(
    const float* __restrict__ x, const int* __restrict__ minidx,
    const float* __restrict__ noder, const float* __restrict__ fb0,
    const float* __restrict__ fb1, const float* __restrict__ fb2,
    const float* __restrict__ fb3, const _Float16* __restrict__ w16,
    float* __restrict__ pooled, float* __restrict__ feat0) {
  __shared__ __align__(16) _Float16 actX[64 * 32];    // x_dec padded to 32 ch
  __shared__ __align__(16) _Float16 actH1[64 * 64];
  __shared__ __align__(16) _Float16 actH2[64 * 64];
  __shared__ __align__(16) _Float16 actC[64 * 160];   // concat(x,h3) padded
  __shared__ int segs[64];

  int tile = blockIdx.x;         // 0..191
  int br = blockIdx.y;           // 0..15
  int b = br >> 2, r = br & 3;
  int tid = threadIdx.x, wave = tid >> 5, lane = tid & 31;

  for (int i = tid; i < 64 * 32; i += 256) actX[i] = (_Float16)0.f;
  for (int i = tid; i < 64 * 160; i += 256) actC[i] = (_Float16)0.f;
  __syncthreads();

  if (tid < 64) {
    int pt = tid;
    int kn = tile * 64 + pt;
    int n = kn & (Nc - 1);
    int m = minidx[b * KNc + kn];
    segs[pt] = m;
    float x0 = x[(b * 2 + 0) * Nc + n], x1 = x[(b * 2 + 1) * Nc + n];
    float th = 6.283185307179586f * (float)r * 0.25f;
    float ct = __cosf(th), st = __sinf(th);
    float xr0 = ct * x0 - st * x1, xr1 = st * x0 + ct * x1;
    float cx = noder[(br * 2 + 0) * Mc + m];
    float cy = noder[(br * 2 + 1) * Mc + m];
    float d0 = xr0 - cx, d1 = xr1 - cy;
    actX[pt * 32 + 0] = (_Float16)d0;
    actX[pt * 32 + 1] = (_Float16)d1;
    actC[pt * 160 + 0] = (_Float16)d0;   // concat channels 0..1
    actC[pt * 160 + 1] = (_Float16)d1;
  }
  __syncthreads();

  // L1: 64 <- 2(pad 32)
  for (int t = wave; t < 16; t += 8) {
    int tm = t >> 2, tn = t & 3;
    v8f c = {};
    c = tile_mm(w16 + OFF_FW0, 32, tm * 16, actX, 32, tn * 16, 1, 0, 0, c);
    store_relu(c, fb0, actH1, 64, tn * 16, tm * 16, 0);
  }
  __syncthreads();
  // L2: 64 <- 64
  for (int t = wave; t < 16; t += 8) {
    int tm = t >> 2, tn = t & 3;
    v8f c = {};
    c = tile_mm(w16 + OFF_FW1, 64, tm * 16, actH1, 64, tn * 16, 2, 0, 0, c);
    store_relu(c, fb1, actH2, 64, tn * 16, tm * 16, 0);
  }
  __syncthreads();
  // L3: 128 <- 64, write into concat buffer at channel offset 2
  for (int t = wave; t < 32; t += 8) {
    int tm = t >> 2, tn = t & 3;
    v8f c = {};
    c = tile_mm(w16 + OFF_FW2, 64, tm * 16, actH2, 64, tn * 16, 2, 0, 0, c);
    store_relu(c, fb2, actC, 160, tn * 16, tm * 16, 2);
  }
  __syncthreads();
  // L4: 128 <- 130(pad 160), fused segment max into pooled
  for (int t = wave; t < 32; t += 8) {
    int tm = t >> 2, tn = t & 3;
    v8f c = {};
    c = tile_mm(w16 + OFF_FW3, 160, tm * 16, actC, 160, tn * 16, 5, 0, 0, c);
    int nn = lane & 15, hi = lane >> 4;
    int pt = tn * 16 + nn;
    int seg = segs[pt];
    int kn = tile * 64 + pt;
#pragma unroll
    for (int i = 0; i < 8; ++i) {
      int oc = tm * 16 + hi * 8 + i;
      float v = c[i] + fb3[oc];
      v = v > 0.f ? v : 0.f;
      // post-ReLU values are >= 0, so int-max on float bits == float max
      atomicMax((int*)&pooled[(br * 128 + oc) * Mc + seg], __float_as_int(v));
      if (kn == 0) feat0[br * 128 + oc] = v;
    }
  }
}

// ---------------- fused gp PointResNet over nodes + global max ---------------
// grid: (M/16, B*R), block 256. 16-node tile.
__global__ __launch_bounds__(256) void gp_kernel(
    const float* __restrict__ noder, const float* __restrict__ pooled,
    const float* __restrict__ feat0, const int* __restrict__ rowmax,
    const float* __restrict__ gb0, const float* __restrict__ gb1,
    const float* __restrict__ gb2, const float* __restrict__ gb3,
    const _Float16* __restrict__ w16, float* __restrict__ out) {
  __shared__ __align__(16) _Float16 actG[16 * 160];   // g_in padded 130->160
  __shared__ __align__(16) _Float16 gH3[16 * 512];    // first 16*256 doubles as h1
  __shared__ __align__(16) _Float16 gH2[16 * 256];
  _Float16* gH1 = gH3;  // h1 dead before h3 is produced

  int ntile = blockIdx.x;  // 0..3
  int br = blockIdx.y;
  int b = br >> 2;
  int tid = threadIdx.x, wave = tid >> 5, lane = tid & 31;

  for (int i = tid; i < 16 * 160; i += 256) {
    int pt = i / 160, ch = i % 160;
    int m = ntile * 16 + pt;
    float v = 0.f;
    if (ch < 2) {
      v = noder[(br * 2 + ch) * Mc + m];
    } else if (ch < 130) {
      int cc = ch - 2;
      v = rowmax[b * Mc + m] ? pooled[(br * 128 + cc) * Mc + m]
                             : feat0[br * 128 + cc];
    }
    actG[i] = (_Float16)v;
  }
  __syncthreads();

  // L1: 256 <- 130(pad 160)
  for (int t = wave; t < 16; t += 8) {
    v8f c = {};
    c = tile_mm(w16 + OFF_GW0, 160, t * 16, actG, 160, 0, 5, 0, 0, c);
    store_relu(c, gb0, gH1, 256, 0, t * 16, 0);
  }
  __syncthreads();
  // L2: 256 <- 256
  for (int t = wave; t < 16; t += 8) {
    v8f c = {};
    c = tile_mm(w16 + OFF_GW1, 256, t * 16, gH1, 256, 0, 8, 0, 0, c);
    store_relu(c, gb1, gH2, 256, 0, t * 16, 0);
  }
  __syncthreads();
  // L3: 512 <- 256 (overwrites h1 region; h1 is dead)
  for (int t = wave; t < 32; t += 8) {
    v8f c = {};
    c = tile_mm(w16 + OFF_GW2, 256, t * 16, gH2, 256, 0, 8, 0, 0, c);
    store_relu(c, gb2, gH3, 512, 0, t * 16, 0);
  }
  __syncthreads();
  // L4: 512 <- concat(g_in 130, h3 512) padded to 672 (split-aligned weights)
  for (int t = wave; t < 32; t += 8) {
    v8f c = {};
    c = tile_mm(w16 + OFF_GW3, 672, t * 16, actG, 160, 0, 5, 0, 0, c);
    c = tile_mm(w16 + OFF_GW3, 672, t * 16, gH3, 512, 0, 16, 160, 0, c);
    int hi = lane >> 4;
#pragma unroll
    for (int i = 0; i < 8; ++i) {
      int oc = t * 16 + hi * 8 + i;
      float v = c[i] + gb3[oc];
      v = v > 0.f ? v : 0.f;
      // max over nodes (lanes) and rotations (blocks) folded into atomics
      atomicMax((int*)&out[b * Fc + oc], __float_as_int(v));
    }
  }
}

// ---------------- host side --------------------------------------------------

extern "C" void kernel_launch(void* const* d_in, const int* in_sizes, int n_in,
                              void* d_out, int out_size, void* d_ws, size_t ws_size,
                              hipStream_t stream) {
  const float* x     = (const float*)d_in[0];
  const float* node  = (const float*)d_in[2];
  const float* fpW0  = (const float*)d_in[4];
  const float* fpb0  = (const float*)d_in[5];
  const float* fpW1  = (const float*)d_in[6];
  const float* fpb1  = (const float*)d_in[7];
  const float* fpW2  = (const float*)d_in[8];
  const float* fpb2  = (const float*)d_in[9];
  const float* fpW3  = (const float*)d_in[10];
  const float* fpb3  = (const float*)d_in[11];
  const float* gpW0  = (const float*)d_in[12];
  const float* gpb0  = (const float*)d_in[13];
  const float* gpW1  = (const float*)d_in[14];
  const float* gpb1  = (const float*)d_in[15];
  const float* gpW2  = (const float*)d_in[16];
  const float* gpb2  = (const float*)d_in[17];
  const float* gpW3  = (const float*)d_in[18];
  const float* gpb3  = (const float*)d_in[19];
  float* out = (float*)d_out;

  char* wsb = (char*)d_ws;
  float*    csum   = (float*)(wsb + 0);        // 768 f
  float*    noder  = (float*)(wsb + 4096);     // 2048 f
  int*      rowmax = (int*)(wsb + 16384);      // 256 i
  int*      minidx = (int*)(wsb + 20480);      // 49152 i
  float*    pooled = (float*)(wsb + 262144);   // 131072 f
  float*    feat0  = (float*)(wsb + 786432);   // 2048 f
  _Float16* w16    = (_Float16*)(wsb + 802816);// 616448 h

  // zero accumulators (post-ReLU values >= 0, so 0-init works for int-max trick)
  zerof_kernel<<<(768 + 255) / 256, 256, 0, stream>>>(csum, 768);
  zerof_kernel<<<(131072 + 255) / 256, 256, 0, stream>>>(pooled, 131072);
  zerof_kernel<<<(2048 + 255) / 256, 256, 0, stream>>>(out, Bc * Fc);

  // pack weights to padded f16
  auto gw = [](int n) { return (n + 255) / 256; };
  wprep_kernel<<<gw(64 * 32), 256, 0, stream>>>(fpW0, w16 + OFF_FW0, 64, 2, 32, 0, 0);
  wprep_kernel<<<gw(64 * 64), 256, 0, stream>>>(fpW1, w16 + OFF_FW1, 64, 64, 64, 0, 0);
  wprep_kernel<<<gw(128 * 64), 256, 0, stream>>>(fpW2, w16 + OFF_FW2, 128, 64, 64, 0, 0);
  wprep_kernel<<<gw(128 * 160), 256, 0, stream>>>(fpW3, w16 + OFF_FW3, 128, 130, 160, 0, 0);
  wprep_kernel<<<gw(256 * 160), 256, 0, stream>>>(gpW0, w16 + OFF_GW0, 256, 130, 160, 0, 0);
  wprep_kernel<<<gw(256 * 256), 256, 0, stream>>>(gpW1, w16 + OFF_GW1, 256, 256, 256, 0, 0);
  wprep_kernel<<<gw(512 * 256), 256, 0, stream>>>(gpW2, w16 + OFF_GW2, 512, 256, 256, 0, 0);
  wprep_kernel<<<gw(512 * 672), 256, 0, stream>>>(gpW3, w16 + OFF_GW3, 512, 642, 672, 130, 160);

  // top-3 SOM assignment + cluster sums
  topk_kernel<<<(Bc * Nc + 255) / 256, 256, 0, stream>>>(x, node, minidx, csum);
  // cluster means -> rotated nodes
  nodes_kernel<<<1, 256, 0, stream>>>(csum, noder, rowmax);

  // fused fp MLP (WMMA) + scatter-max
  fp_kernel<<<dim3(KNc / 64, Bc * Rc), 256, 0, stream>>>(
      x, minidx, noder, fpb0, fpb1, fpb2, fpb3, w16, pooled, feat0);

  // fused gp MLP (WMMA) + global max over nodes/rotations
  gp_kernel<<<dim3(Mc / 16, Bc * Rc), 256, 0, stream>>>(
      noder, pooled, feat0, rowmax, gpb0, gpb1, gpb2, gpb3, w16, out);

  (void)in_sizes; (void)n_in; (void)out_size; (void)ws_size;
}
